// GraphNet_33122787787019
// MI455X (gfx1250) — compile-verified, matrix operand order checked by reading, not run on hardware
//
#include <hip/hip_runtime.h>
#include <hip/hip_fp16.h>
#include <stdint.h>

// ---------------------------------------------------------------------------
// CDNA5 (gfx1250) GraphNet forward. All 32-wide linear layers are executed as
// v_wmma_f32_16x16x32_f16: M=16 rows (edges/nodes/graphs) per wave, N=32 in
// two 16-wide tiles, K tiled by 32 (zero-padded where din % 32 != 0).
// BatchNorm (eval) is folded into layer-2 weights at prep time.
// ---------------------------------------------------------------------------

typedef __attribute__((ext_vector_type(16))) _Float16 v16h;
typedef __attribute__((ext_vector_type(8)))  float    v8f;

union HFrag { uint4 q[2]; v16h h; };

// A-fragment (16x32 f16, ISA 7.12.2): lane L<16 row M=L holds K=0..7 (v0-3)
// and K=16..23 (v4-7); lane L+16 holds K=8..15 and K=24..31.
__device__ __forceinline__ v16h ld_afrag(const __half* row, int kt, int lane) {
  const __half* p = row + kt * 32 + ((lane & 16) ? 8 : 0);
  HFrag f;
  f.q[0] = *reinterpret_cast<const uint4*>(p);
  f.q[1] = *reinterpret_cast<const uint4*>(p + 16);
  return f.h;
}

// B-fragments are pre-packed by k_prep as [tile][lane][8 dwords].
__device__ __forceinline__ v16h ld_bfrag(const uint32_t* wf, int tile, int lane) {
  const uint4* q = reinterpret_cast<const uint4*>(wf + (size_t)(tile * 32 + lane) * 8);
  HFrag f;
  f.q[0] = q[0];
  f.q[1] = q[1];
  return f.h;
}

__device__ __forceinline__ void st4h(__half* dst, float4 v) {
  *reinterpret_cast<__half2*>(dst)     = __floats2half2_rn(v.x, v.y);
  *reinterpret_cast<__half2*>(dst + 2) = __floats2half2_rn(v.z, v.w);
}

// Two-layer MLP (din->32 ReLU ->32) on one 16-row tile. tA: 16 x DINP halves
// (row-major), tH: 16 x 32 halves scratch. out = two C fragments (f32) with
// layer-2 bias added. C layout: lane owns column n=lane&15, rows
// mb..mb+7 where mb = (lane&16)>>1.
template<int DINP, int KT>
__device__ __forceinline__ void mlp_core(const __half* tA, __half* tH, int lane,
    const uint32_t* __restrict__ wf1, const float* __restrict__ fb1,
    const uint32_t* __restrict__ wf2, const float* __restrict__ fb2,
    v8f out[2]) {
  const __half* rowA = tA + (lane & 15) * DINP;
  const int mb = (lane & 16) >> 1;
  const int nl = lane & 15;
#pragma unroll
  for (int nt = 0; nt < 2; ++nt) {
    v8f c = {};
#pragma unroll
    for (int kt = 0; kt < KT; ++kt) {
      v16h a = ld_afrag(rowA, kt, lane);
      v16h b = ld_bfrag(wf1, nt * KT + kt, lane);
      c = __builtin_amdgcn_wmma_f32_16x16x32_f16(false, a, false, b,
                                                 (short)0, c, false, false);
    }
    float bias = fb1[nt * 16 + nl];
#pragma unroll
    for (int v = 0; v < 8; ++v) {
      float hv = c[v] + bias;
      hv = hv > 0.f ? hv : 0.f;                    // ReLU (BN folded into l2)
      tH[(mb + v) * 32 + nt * 16 + nl] = (__half)hv;
    }
  }
  // Same-wave LDS RAW: DS ops are in-order per wave; explicit wait for safety.
  asm volatile("s_wait_dscnt 0" ::: "memory");
  const __half* rowH = tH + nl * 32;
#pragma unroll
  for (int nt = 0; nt < 2; ++nt) {
    v16h a = ld_afrag(rowH, 0, lane);
    v16h b = ld_bfrag(wf2, nt, lane);
    v8f c = {};
    c = __builtin_amdgcn_wmma_f32_16x16x32_f16(false, a, false, b,
                                               (short)0, c, false, false);
    float bias = fb2[nt * 16 + nl];
#pragma unroll
    for (int v = 0; v < 8; ++v) out[nt][v] = c[v] + bias;
  }
}

// ------------------------------ weight prep --------------------------------
struct MlpDesc {
  const float *w1, *b1, *g, *bt, *rm, *rv, *w2, *b2;
  uint32_t* wf1; float* fb1; uint32_t* wf2; float* fb2;
  int din, dinp;
};
struct PrepArgs { MlpDesc d[8]; };

__global__ __launch_bounds__(256) void k_prep(PrepArgs A) {
  MlpDesc d = A.d[blockIdx.x];
  __shared__ float a_s[32], c_s[32];
  int t = threadIdx.x;
  if (t < 32) {
    float a = 1.f, c = 0.f;
    if (d.g != nullptr) {                       // fold eval BatchNorm
      float inv = rsqrtf(d.rv[t] + 1e-5f);
      a = d.g[t] * inv;
      c = d.bt[t] - d.rm[t] * a;
    }
    a_s[t] = a; c_s[t] = c;
    d.fb1[t] = d.b1[t];
  }
  __syncthreads();
  if (t < 32) {
    float s = d.b2[t];
    for (int k = 0; k < 32; ++k) s += d.w2[t * 32 + k] * c_s[k];
    d.fb2[t] = s;                               // b2' = b2 + W2 c
  }
  const int KT = d.dinp >> 5;
  const int n1 = 2 * KT * 256;                  // dwords in wf1
  for (int i = t; i < n1; i += 256) {
    int v = i & 7, lane = (i >> 3) & 31, tile = i >> 8;
    int kt = tile % KT, nt = tile / KT;
    int n  = nt * 16 + (lane & 15);
    int kb = kt * 32 + ((v < 4) ? 0 : 16) + ((lane < 16) ? 0 : 8) + 2 * (v & 3);
    float f0 = (kb     < d.din) ? d.w1[n * d.din + kb]     : 0.f;
    float f1 = (kb + 1 < d.din) ? d.w1[n * d.din + kb + 1] : 0.f;
    __half2 h = __floats2half2_rn(f0, f1);
    d.wf1[i] = *reinterpret_cast<uint32_t*>(&h);
  }
  for (int i = t; i < 512; i += 256) {          // wf2 (K=32), scaled by a
    int v = i & 7, lane = (i >> 3) & 31, nt = i >> 8;
    int n  = nt * 16 + (lane & 15);
    int kb = ((v < 4) ? 0 : 16) + ((lane < 16) ? 0 : 8) + 2 * (v & 3);
    float f0 = d.w2[n * 32 + kb]     * a_s[kb];
    float f1 = d.w2[n * 32 + kb + 1] * a_s[kb + 1];
    __half2 h = __floats2half2_rn(f0, f1);
    d.wf2[i] = *reinterpret_cast<uint32_t*>(&h);
  }
}

// ------------------------------ edge block ---------------------------------
template<int DINP, int KT, int DE, int DU>
__global__ __launch_bounds__(256) void k_edge(
    const float* __restrict__ x, const float* __restrict__ ea,
    const float* __restrict__ u, const int* __restrict__ row,
    const int* __restrict__ col, const int* __restrict__ batch,
    const uint32_t* __restrict__ wf1, const float* __restrict__ fb1,
    const uint32_t* __restrict__ wf2, const float* __restrict__ fb2,
    float* __restrict__ eout, int E_) {
  __shared__ __half sA[8 * 16 * DINP];
  __shared__ __half sH[8 * 16 * 32];
  const int wave = threadIdx.x >> 5, lane = threadIdx.x & 31;
  __half* tA = sA + wave * 16 * DINP;
  __half* tH = sH + wave * 16 * 32;
  const long tbase = ((long)blockIdx.x * 8 + wave) * 16;
  if (tbase >= E_) return;
  const int nl = lane & 15;
  long e = tbase + nl; if (e >= E_) e = E_ - 1;
  __half* dst = tA + nl * DINP;
  if (lane < 16) {                                   // x[row] | edge_attr
    const float4* xs = reinterpret_cast<const float4*>(x + (long)row[e] * 32);
#pragma unroll
    for (int q = 0; q < 8; ++q) st4h(dst + q * 4, xs[q]);
    const float4* es = reinterpret_cast<const float4*>(ea + e * DE);
#pragma unroll
    for (int q = 0; q < DE / 4; ++q) st4h(dst + 64 + q * 4, es[q]);
  } else {                                           // x[col] | u[batch[row]]
    const float4* xs = reinterpret_cast<const float4*>(x + (long)col[e] * 32);
#pragma unroll
    for (int q = 0; q < 8; ++q) st4h(dst + 32 + q * 4, xs[q]);
    const float4* us =
        reinterpret_cast<const float4*>(u + (long)batch[row[e]] * DU);
#pragma unroll
    for (int q = 0; q < DU / 4; ++q) st4h(dst + 64 + DE + q * 4, us[q]);
  }
  v8f out[2];
  mlp_core<DINP, KT>(tA, tH, lane, wf1, fb1, wf2, fb2, out);
  const int mb = (lane & 16) >> 1;
#pragma unroll
  for (int nt = 0; nt < 2; ++nt)
#pragma unroll
    for (int v = 0; v < 8; ++v) {
      long ee = tbase + mb + v;
      if (ee < E_) eout[ee * 32 + nt * 16 + nl] = out[nt][v];
    }
}

// --------------------- node block 1 (messages + scatter) -------------------
__global__ __launch_bounds__(256) void k_msg(
    const float* __restrict__ x, const float* __restrict__ ea,
    const int* __restrict__ row, const int* __restrict__ col,
    const uint32_t* __restrict__ wf1, const float* __restrict__ fb1,
    const uint32_t* __restrict__ wf2, const float* __restrict__ fb2,
    float* __restrict__ agg, float* __restrict__ cnt, int E_) {
  __shared__ __half sA[8 * 16 * 64];
  __shared__ __half sH[8 * 16 * 32];
  const int wave = threadIdx.x >> 5, lane = threadIdx.x & 31;
  __half* tA = sA + wave * 16 * 64;
  __half* tH = sH + wave * 16 * 32;
  const long tbase = ((long)blockIdx.x * 8 + wave) * 16;
  if (tbase >= E_) return;
  const int nl = lane & 15;
  long e = tbase + nl; if (e >= E_) e = E_ - 1;
  __half* dst = tA + nl * 64;
  if (lane < 16) {
    const float4* xs = reinterpret_cast<const float4*>(x + (long)row[e] * 32);
#pragma unroll
    for (int q = 0; q < 8; ++q) st4h(dst + q * 4, xs[q]);
  } else {
    const float4* es = reinterpret_cast<const float4*>(ea + e * 32);
#pragma unroll
    for (int q = 0; q < 8; ++q) st4h(dst + 32 + q * 4, es[q]);
  }
  v8f out[2];
  mlp_core<64, 2>(tA, tH, lane, wf1, fb1, wf2, fb2, out);
  const int mb = (lane & 16) >> 1;
#pragma unroll
  for (int nt = 0; nt < 2; ++nt)
#pragma unroll
    for (int v = 0; v < 8; ++v) {
      long ee = tbase + mb + v;
      if (ee < E_)
        atomicAdd(&agg[(long)col[ee] * 32 + nt * 16 + nl], out[nt][v]);
    }
  if (lane < 16 && tbase + nl < E_)
    atomicAdd(&cnt[col[tbase + nl]], 1.0f);
}

// ----------------------------- node block 2 --------------------------------
template<int DU>
__global__ __launch_bounds__(256) void k_node2(
    const float* __restrict__ x, const float* __restrict__ agg,
    const float* __restrict__ cnt, const float* __restrict__ u,
    const int* __restrict__ batch,
    const uint32_t* __restrict__ wf1, const float* __restrict__ fb1,
    const uint32_t* __restrict__ wf2, const float* __restrict__ fb2,
    float* __restrict__ xout, float* __restrict__ gagg,
    float* __restrict__ gcnt, int N_) {
  constexpr int DINP = 96;
  __shared__ __half sA[8 * 16 * DINP];
  __shared__ __half sH[8 * 16 * 32];
  const int wave = threadIdx.x >> 5, lane = threadIdx.x & 31;
  __half* tA = sA + wave * 16 * DINP;
  __half* tH = sH + wave * 16 * 32;
  const long tbase = ((long)blockIdx.x * 8 + wave) * 16;
  if (tbase >= N_) return;
  const int nl = lane & 15;
  long i = tbase + nl; if (i >= N_) i = N_ - 1;
  __half* dst = tA + nl * DINP;
  if (lane < 16) {                                   // x | agg/cnt
    const float4* xs = reinterpret_cast<const float4*>(x + i * 32);
#pragma unroll
    for (int q = 0; q < 8; ++q) st4h(dst + q * 4, xs[q]);
    float inv = 1.f / fmaxf(cnt[i], 1.f);
    const float4* as = reinterpret_cast<const float4*>(agg + i * 32);
#pragma unroll
    for (int q = 0; q < 8; ++q) {
      float4 v = as[q];
      v.x *= inv; v.y *= inv; v.z *= inv; v.w *= inv;
      st4h(dst + 32 + q * 4, v);
    }
  } else {                                           // u[batch] (+ zero pad)
    const float4* us = reinterpret_cast<const float4*>(u + (long)batch[i] * DU);
#pragma unroll
    for (int q = 0; q < DU / 4; ++q) st4h(dst + 64 + q * 4, us[q]);
    if (DU == 16) {
      float4 z = {0.f, 0.f, 0.f, 0.f};
#pragma unroll
      for (int q = 0; q < 4; ++q) st4h(dst + 80 + q * 4, z);
    }
  }
  v8f out[2];
  mlp_core<DINP, 3>(tA, tH, lane, wf1, fb1, wf2, fb2, out);
  const int mb = (lane & 16) >> 1;
#pragma unroll
  for (int nt = 0; nt < 2; ++nt)
#pragma unroll
    for (int v = 0; v < 8; ++v) {
      long nn = tbase + mb + v;
      if (nn < N_) {
        float val = out[nt][v];
        xout[nn * 32 + nt * 16 + nl] = val;
        atomicAdd(&gagg[(long)batch[nn] * 32 + nt * 16 + nl], val);
      }
    }
  if (lane < 16 && tbase + nl < N_)
    atomicAdd(&gcnt[batch[tbase + nl]], 1.0f);
}

// ----------------------------- global block --------------------------------
template<int DU>
__global__ __launch_bounds__(256) void k_glob(
    const float* __restrict__ u, const float* __restrict__ gagg,
    const float* __restrict__ gcnt,
    const uint32_t* __restrict__ wf1, const float* __restrict__ fb1,
    const uint32_t* __restrict__ wf2, const float* __restrict__ fb2,
    float* __restrict__ uout, int G_) {
  constexpr int DINP = 64;
  __shared__ __half sA[8 * 16 * DINP];
  __shared__ __half sH[8 * 16 * 32];
  const int wave = threadIdx.x >> 5, lane = threadIdx.x & 31;
  __half* tA = sA + wave * 16 * DINP;
  __half* tH = sH + wave * 16 * 32;
  const long tbase = ((long)blockIdx.x * 8 + wave) * 16;
  if (tbase >= G_) return;
  const int nl = lane & 15;
  long g = tbase + nl; if (g >= G_) g = G_ - 1;
  __half* dst = tA + nl * DINP;
  if (lane < 16) {                                   // u
    const float4* us = reinterpret_cast<const float4*>(u + g * DU);
#pragma unroll
    for (int q = 0; q < DU / 4; ++q) st4h(dst + q * 4, us[q]);
  } else {                                           // gagg/gcnt (+ pad)
    float inv = 1.f / fmaxf(gcnt[g], 1.f);
    const float4* as = reinterpret_cast<const float4*>(gagg + g * 32);
#pragma unroll
    for (int q = 0; q < 8; ++q) {
      float4 v = as[q];
      v.x *= inv; v.y *= inv; v.z *= inv; v.w *= inv;
      st4h(dst + DU + q * 4, v);
    }
    if (DU == 16) {
      float4 z = {0.f, 0.f, 0.f, 0.f};
#pragma unroll
      for (int q = 0; q < 4; ++q) st4h(dst + 48 + q * 4, z);
    }
  }
  v8f out[2];
  mlp_core<DINP, 2>(tA, tH, lane, wf1, fb1, wf2, fb2, out);
  const int mb = (lane & 16) >> 1;
#pragma unroll
  for (int nt = 0; nt < 2; ++nt)
#pragma unroll
    for (int v = 0; v < 8; ++v) {
      long gg = tbase + mb + v;
      if (gg < G_) uout[gg * 32 + nt * 16 + nl] = out[nt][v];
    }
}

// ------------------------- classifier + log_softmax ------------------------
__global__ __launch_bounds__(256) void k_cls(
    const float* __restrict__ x, const float* __restrict__ W,
    const float* __restrict__ b, float* __restrict__ out, int N_) {
  __shared__ float sW[640];
  __shared__ float sB[20];
  int t = threadIdx.x;
  for (int i = t; i < 640; i += 256) sW[i] = W[i];
  if (t < 20) sB[t] = b[t];
  __syncthreads();
  long i = (long)blockIdx.x * 256 + t;
  if (i >= N_) return;
  float xi[32];
  const float4* xr = reinterpret_cast<const float4*>(x + i * 32);
#pragma unroll
  for (int q = 0; q < 8; ++q) {
    float4 v = xr[q];
    xi[4 * q] = v.x; xi[4 * q + 1] = v.y; xi[4 * q + 2] = v.z; xi[4 * q + 3] = v.w;
  }
  float lg[20]; float mx = -1e30f;
#pragma unroll
  for (int o = 0; o < 20; ++o) {
    float s = sB[o];
#pragma unroll
    for (int k = 0; k < 32; ++k) s += sW[o * 32 + k] * xi[k];
    lg[o] = s; mx = fmaxf(mx, s);
  }
  float se = 0.f;
#pragma unroll
  for (int o = 0; o < 20; ++o) se += __expf(lg[o] - mx);
  float ls = mx + __logf(se);
#pragma unroll
  for (int o = 0; o < 20; ++o) out[i * 20 + o] = lg[o] - ls;
}

__global__ void k_zero(float* p, size_t n) {
  size_t i = (size_t)blockIdx.x * blockDim.x + threadIdx.x;
  if (i < n) p[i] = 0.f;
}

// ------------------------------- host side ---------------------------------
extern "C" void kernel_launch(void* const* d_in, const int* in_sizes, int n_in,
                              void* d_out, int out_size, void* d_ws, size_t ws_size,
                              hipStream_t stream) {
  (void)out_size; (void)ws_size;
  const int N = in_sizes[0] / 32;   // x: (N,32)
  const int E = in_sizes[1] / 16;   // edge_attr: (E,16)
  const int G = in_sizes[2] / 16;   // u: (G,16)

  // Param leaves in jax pytree (sorted-key) order:
  // cls{b,w}; gn1{edge(bt,g,l1.b,l1.w,l2.b,l2.w,rm,rv), glob(l1.b,l1.w,l2.b,l2.w),
  // node1(8), node2(8)}; gn2{same}.
  static const int leaf_sizes[58] = {
    20, 640,
    32,32,32, 32*96,  32, 1024, 32,32,   // gn1.edge
    32, 32*48, 32, 1024,                 // gn1.glob
    32,32,32, 32*64,  32, 1024, 32,32,   // gn1.node1
    32,32,32, 32*80,  32, 1024, 32,32,   // gn1.node2
    32,32,32, 32*128, 32, 1024, 32,32,   // gn2.edge
    32, 32*64, 32, 1024,                 // gn2.glob
    32,32,32, 32*64,  32, 1024, 32,32,   // gn2.node1
    32,32,32, 32*96,  32, 1024, 32,32,   // gn2.node2
  };
  const float* P[58];
  const int *ei, *batch;
  if (n_in >= 63) {                         // fully flattened leaves
    for (int i = 0; i < 58; ++i) P[i] = (const float*)d_in[3 + i];
    ei = (const int*)d_in[61]; batch = (const int*)d_in[62];
  } else {                                  // params as one contiguous blob
    const float* blob = (const float*)d_in[3];
    size_t off = 0;
    for (int i = 0; i < 58; ++i) { P[i] = blob + off; off += leaf_sizes[i]; }
    ei = (const int*)d_in[4]; batch = (const int*)d_in[5];
  }
  const int* row = ei;
  const int* col = ei + E;
  const float* x0 = (const float*)d_in[0];
  const float* e0 = (const float*)d_in[1];
  const float* u0 = (const float*)d_in[2];

  float* out_ls = (float*)d_out;
  float* x_fin  = out_ls + (size_t)N * 20;
  float* e_fin  = x_fin + (size_t)N * 32;
  float* u_fin  = e_fin + (size_t)E * 32;

  char* ws = (char*)d_ws;
  size_t off = 0;
  auto take = [&](size_t bytes) -> void* {
    void* p = ws + off;
    off = (off + bytes + 255) & ~(size_t)255;
    return p;
  };
  float* e1buf = (float*)take((size_t)E * 32 * 4);
  float* x1    = (float*)take((size_t)N * 32 * 4);
  float* u1    = (float*)take((size_t)G * 32 * 4);
  size_t aggOff = off;
  float* agg   = (float*)take((size_t)N * 32 * 4);
  float* cnt   = (float*)take((size_t)N * 4);
  float* gagg  = (float*)take((size_t)G * 32 * 4);
  float* gcnt  = (float*)take((size_t)G * 4);
  size_t zeroFloats = (off - aggOff) / 4;
  float* zbase = (float*)(ws + aggOff);

  // 8 MLPs: e1, n1_1, n2_1, g1, e2, n1_2, n2_2, g2
  const int  mlp_base[8] = {2, 14, 22, 10, 30, 42, 50, 38};
  const bool is_bn[8]    = {true, true, true, false, true, true, true, false};
  const int  dins[8]     = {96, 64, 80, 48, 128, 64, 96, 64};
  const int  dinps[8]    = {96, 64, 96, 64, 128, 64, 96, 64};
  PrepArgs PA;
  struct FragPtrs { uint32_t* wf1; float* fb1; uint32_t* wf2; float* fb2; } FP[8];
  for (int m = 0; m < 8; ++m) {
    int b = mlp_base[m];
    MlpDesc& d = PA.d[m];
    if (is_bn[m]) {
      d.bt = P[b];     d.g  = P[b + 1]; d.b1 = P[b + 2]; d.w1 = P[b + 3];
      d.b2 = P[b + 4]; d.w2 = P[b + 5]; d.rm = P[b + 6]; d.rv = P[b + 7];
    } else {
      d.b1 = P[b]; d.w1 = P[b + 1]; d.b2 = P[b + 2]; d.w2 = P[b + 3];
      d.g = nullptr; d.bt = nullptr; d.rm = nullptr; d.rv = nullptr;
    }
    d.din = dins[m]; d.dinp = dinps[m];
    int KT = dinps[m] / 32;
    d.wf1 = (uint32_t*)take((size_t)2 * KT * 256 * 4);
    d.fb1 = (float*)take(128);
    d.wf2 = (uint32_t*)take(512 * 4);
    d.fb2 = (float*)take(128);
    FP[m] = {d.wf1, d.fb1, d.wf2, d.fb2};
  }

  k_prep<<<8, 256, 0, stream>>>(PA);

  const int tilesE = (E + 15) / 16, blkE = (tilesE + 7) / 8;
  const int tilesN = (N + 15) / 16, blkN = (tilesN + 7) / 8;
  const int tilesG = (G + 15) / 16, blkG = (tilesG + 7) / 8;
  const int blkZ = (int)((zeroFloats + 255) / 256);

  // ---------------- metalayer 1 ----------------
  k_zero<<<blkZ, 256, 0, stream>>>(zbase, zeroFloats);
  k_edge<96, 3, 16, 16><<<blkE, 256, 0, stream>>>(x0, e0, u0, row, col, batch,
      FP[0].wf1, FP[0].fb1, FP[0].wf2, FP[0].fb2, e1buf, E);
  k_msg<<<blkE, 256, 0, stream>>>(x0, e1buf, row, col,
      FP[1].wf1, FP[1].fb1, FP[1].wf2, FP[1].fb2, agg, cnt, E);
  k_node2<16><<<blkN, 256, 0, stream>>>(x0, agg, cnt, u0, batch,
      FP[2].wf1, FP[2].fb1, FP[2].wf2, FP[2].fb2, x1, gagg, gcnt, N);
  k_glob<16><<<blkG, 256, 0, stream>>>(u0, gagg, gcnt,
      FP[3].wf1, FP[3].fb1, FP[3].wf2, FP[3].fb2, u1, G);

  // ---------------- metalayer 2 ----------------
  k_zero<<<blkZ, 256, 0, stream>>>(zbase, zeroFloats);
  k_edge<128, 4, 32, 32><<<blkE, 256, 0, stream>>>(x1, e1buf, u1, row, col, batch,
      FP[4].wf1, FP[4].fb1, FP[4].wf2, FP[4].fb2, e_fin, E);
  k_msg<<<blkE, 256, 0, stream>>>(x1, e_fin, row, col,
      FP[5].wf1, FP[5].fb1, FP[5].wf2, FP[5].fb2, agg, cnt, E);
  k_node2<32><<<blkN, 256, 0, stream>>>(x1, agg, cnt, u1, batch,
      FP[6].wf1, FP[6].fb1, FP[6].wf2, FP[6].fb2, x_fin, gagg, gcnt, N);
  k_glob<32><<<blkG, 256, 0, stream>>>(u1, gagg, gcnt,
      FP[7].wf1, FP[7].fb1, FP[7].wf2, FP[7].fb2, u_fin, G);

  // ---------------- classifier ----------------
  k_cls<<<(N + 255) / 256, 256, 0, stream>>>(x_fin, P[1], P[0], out_ls, N);
}
